// UnsupervisedLoss_15324443312685
// MI455X (gfx1250) — matrix operand order repaired and checked by reading, not compiled, not run on hardware
//
#include <hip/hip_runtime.h>
#include <hip/hip_bf16.h>
#include <math.h>

typedef __attribute__((ext_vector_type(16))) _Float16 v16h;
typedef __attribute__((ext_vector_type(8)))  float    v8f;

#define B_ 4
#define P_ 19248
#define C_ 81
#define M_ 32
#define TOPKC 100
#define TOPKI 15
#define PH 138
#define PW 138
#define PPIX (PH*PW)      /* 19044 */
#define OH 550
#define OW 550
#define OPIX (OH*OW)      /* 302500 */
#define EPSF 1e-6f
#define NEGINF (-3.4e38f)

/* ---- workspace layout (float units); total ~1.80M floats (~6.9 MB) ---- */
#define OFF_CONF1   0
#define OFF_TOPKI   (OFF_CONF1 + B_*P_)          /* 400 ints   */
#define OFF_TOPKV   (OFF_TOPKI + 400)            /* 400 f      */
#define OFF_SLOC    (OFF_TOPKV + 400)            /* 1600 f     */
#define OFF_SMASK   (OFF_SLOC + 1600)            /* 12800 f    */
#define OFF_GAUSS   (OFF_SMASK + 12800)          /* 409600 f   */
#define OFF_IOUMAX  (OFF_GAUSS + 409600)         /* 400 u32    */
#define OFF_KCX     (OFF_IOUMAX + 400)           /* 64 f       */
#define OFF_KCY     (OFF_KCX + 64)
#define OFF_KSX     (OFF_KCY + 64)
#define OFF_KSY     (OFF_KSX + 64)
#define OFF_KCONF   (OFF_KSY + 64)
#define OFF_KMASKW  (OFF_KCONF + 64)             /* 2048 f     */
#define OFF_FINAL   (OFF_KMASKW + 2048)          /* 76176 f    */
#define OFF_RESIZED (OFF_FINAL + PPIX*B_)        /* 1210000 f  */
#define OFF_PART    (OFF_RESIZED + (size_t)B_*OPIX)

__device__ __forceinline__ float fast_rcp(float x) { return __builtin_amdgcn_rcpf(x); }

/* ---------------- 1. softmax class-1 probability, one wave per row ---------------- */
__global__ void k_conf1(const float* __restrict__ conf, float* __restrict__ conf1) {
  int gid  = blockIdx.x * blockDim.x + threadIdx.x;
  int wave = gid >> 5, lane = gid & 31;
  if (wave >= B_ * P_) return;
  const float* row = conf + (size_t)wave * C_;
  float v0 = row[lane];
  float v1 = row[lane + 32];
  float v2 = (lane + 64 < C_) ? row[lane + 64] : NEGINF;
  float m  = fmaxf(v0, fmaxf(v1, v2));
  #pragma unroll
  for (int d = 16; d >= 1; d >>= 1) m = fmaxf(m, __shfl_xor(m, d, 32));
  float s = __expf(v0 - m) + __expf(v1 - m) + ((lane + 64 < C_) ? __expf(v2 - m) : 0.f);
  #pragma unroll
  for (int d = 16; d >= 1; d >>= 1) s += __shfl_xor(s, d, 32);
  if (lane == 0) conf1[wave] = __expf(row[1] - m) * fast_rcp(s);
}

/* ---------------- 2. top-100 per batch: wave-shuffle argmax, 2 barriers/iter ---------------- */
__global__ void k_topk(float* __restrict__ conf1, int* __restrict__ tki, float* __restrict__ tkv) {
  __shared__ float wv[32];
  __shared__ int   wi[32];
  int b = blockIdx.x, tid = threadIdx.x;          /* 1024 threads = 32 waves */
  int lane = tid & 31, wid = tid >> 5;
  float* base = conf1 + (size_t)b * P_;
  for (int t = 0; t < TOPKC; ++t) {
    float best = NEGINF; int bi = 0x7fffffff;
    for (int p = tid; p < P_; p += 1024) {
      float v = base[p];
      if (v > best) { best = v; bi = p; }         /* strided scan: strict > keeps lowest p */
    }
    #pragma unroll
    for (int d = 16; d >= 1; d >>= 1) {           /* wave argmax, stable tie-break */
      float ov = __shfl_xor(best, d, 32);
      int   oi = __shfl_xor(bi,   d, 32);
      if (ov > best || (ov == best && oi < bi)) { best = ov; bi = oi; }
    }
    if (lane == 0) { wv[wid] = best; wi[wid] = bi; }
    __syncthreads();
    if (wid == 0) {
      float v2 = wv[lane]; int i2 = wi[lane];
      #pragma unroll
      for (int d = 16; d >= 1; d >>= 1) {
        float ov = __shfl_xor(v2, d, 32);
        int   oi = __shfl_xor(i2, d, 32);
        if (ov > v2 || (ov == v2 && oi < i2)) { v2 = ov; i2 = oi; }
      }
      if (lane == 0) {
        tki[b * TOPKC + t] = i2;
        tkv[b * TOPKC + t] = v2;
        base[i2] = NEGINF;                        /* mark taken; conf1 regenerated each call */
      }
    }
    __syncthreads();
  }
}

/* ---------------- 3. 32x32 gaussians + gather sloc/smask, init ioumax ---------------- */
__global__ void k_gauss(const float* __restrict__ loc, const float* __restrict__ mask,
                        const int* __restrict__ tki,
                        float* __restrict__ sloc, float* __restrict__ smask,
                        float* __restrict__ gauss, unsigned* __restrict__ ioumax) {
  int bk = blockIdx.x;                 /* b*100 + k */
  int b = bk / TOPKC, k = bk % TOPKC;
  int tid = threadIdx.x;               /* 256 */
  int idx = tki[bk];
  const float* lp = loc + ((size_t)b * P_ + idx) * 4;
  float cx = lp[0], cy = lp[1];
  float sx = fabsf(lp[2]) * 0.5f + 0.001f;
  float sy = fabsf(lp[3]) * 0.5f + 0.001f;
  float i2sx = fast_rcp(2.f * sx * sx), i2sy = fast_rcp(2.f * sy * sy);
  float* g = gauss + (size_t)bk * 1024;
  for (int p = tid; p < 1024; p += 256) {
    int y = p >> 5, x = p & 31;
    float xs = (x + 0.5f) * 0.03125f, ys = (y + 0.5f) * 0.03125f;
    float dx = xs - cx, dy = ys - cy;
    g[p] = __expf(-(dy * dy * i2sy + dx * dx * i2sx));
  }
  if (tid < 4)  sloc[bk * 4 + tid]  = lp[tid];
  if (tid < 32) smask[bk * 32 + tid] = mask[((size_t)b * P_ + idx) * 32 + tid];
  if (k == 0 && tid < TOPKC) ioumax[b * TOPKC + tid] = 0u;  /* +0.0f bits */
}

/* ---------------- 4. pairwise gaussian IoU, wave per (i<j) pair ---------------- */
__global__ void k_iou(const float* __restrict__ gauss, unsigned* __restrict__ ioumax) {
  const int PAIRS = TOPKC * (TOPKC - 1) / 2;  /* 4950 */
  int gid = blockIdx.x * blockDim.x + threadIdx.x;
  int wave = gid >> 5, lane = gid & 31;
  if (wave >= B_ * PAIRS) return;
  int b = wave / PAIRS, t = wave % PAIRS;
  int i = 0, rem = t;
  while (rem >= TOPKC - 1 - i) { rem -= TOPKC - 1 - i; ++i; }
  int j = i + 1 + rem;
  const float* gi = gauss + (size_t)(b * TOPKC + i) * 1024;
  const float* gj = gauss + (size_t)(b * TOPKC + j) * 1024;
  float inter = 0.f, uni = 0.f;
  for (int p = lane; p < 1024; p += 32) {
    float a = gi[p], c = gj[p];
    inter += fminf(a, c);
    uni   += fmaxf(a, c);
  }
  #pragma unroll
  for (int d = 16; d >= 1; d >>= 1) {
    inter += __shfl_xor(inter, d, 32);
    uni   += __shfl_xor(uni,   d, 32);
  }
  if (lane == 0) {
    float iou = inter / uni;   /* positive => uint bit-pattern order == float order */
    atomicMax(&ioumax[b * TOPKC + j], __float_as_uint(iou));
  }
}

/* ---------------- 5. keep 15 smallest iou_max; stage WMMA operands ---------------- */
__global__ void k_keep(const unsigned* __restrict__ ioumax, const float* __restrict__ tkv,
                       const float* __restrict__ sloc, const float* __restrict__ smask,
                       float* __restrict__ kcx, float* __restrict__ kcy,
                       float* __restrict__ ksx, float* __restrict__ ksy,
                       float* __restrict__ kconf, float* __restrict__ kmaskw) {
  __shared__ float s_iou[TOPKC];
  __shared__ int   s_keep[16];
  int b = blockIdx.x, tid = threadIdx.x;   /* 512 */
  if (tid < TOPKC) s_iou[tid] = __uint_as_float(ioumax[b * TOPKC + tid]);
  __syncthreads();
  if (tid == 0) {
    for (int t = 0; t < TOPKI; ++t) {        /* stable: strict-less keeps lowest index */
      float best = 3.4e38f; int bi = 0;
      for (int j = 0; j < TOPKC; ++j) { float v = s_iou[j]; if (v < best) { best = v; bi = j; } }
      s_keep[t] = bi; s_iou[bi] = 3.4e38f;
    }
    s_keep[15] = 0;
  }
  __syncthreads();
  if (tid < 16) {
    if (tid < TOPKI) {
      int kk = s_keep[tid];
      const float* lp = sloc + (b * TOPKC + kk) * 4;
      kcx[b * 16 + tid] = lp[0];
      kcy[b * 16 + tid] = lp[1];
      ksx[b * 16 + tid] = fabsf(lp[2]) * 0.5f + 0.001f;
      ksy[b * 16 + tid] = fabsf(lp[3]) * 0.5f + 0.001f;
      kconf[b * 16 + tid] = tkv[b * TOPKC + kk];
    } else {
      kcx[b * 16 + 15] = 0.f; kcy[b * 16 + 15] = 0.f;
      ksx[b * 16 + 15] = 1.f; ksy[b * 16 + 15] = 1.f;
      kconf[b * 16 + 15] = 0.f;
    }
  }
  __syncthreads();
  for (int e = tid; e < 16 * 32; e += blockDim.x) {
    int row = e >> 5, ch = e & 31;
    float v = (row < TOPKI) ? smask[(size_t)(b * TOPKC + s_keep[row]) * 32 + ch] : 0.f;
    kmaskw[((size_t)b * 16 + row) * 32 + ch] = v;   /* B operand, zero-padded col 15 */
  }
}

/* ---------------- 6. WMMA assemble + fused sigmoid/gaussian/final_conf ----------------
 * One wave per 16-pixel tile: D(16x16 f32) = protoTile(16x32 f16) x kmask^T(32x16 f16)
 * via a single V_WMMA_F32_16X16X32_F16, then epilogue reduces over k in-register. */
__global__ void k_assemble(const float* __restrict__ proto, const float* __restrict__ kmaskw,
                           const float* __restrict__ kcx, const float* __restrict__ kcy,
                           const float* __restrict__ ksx, const float* __restrict__ ksy,
                           const float* __restrict__ kconf, float* __restrict__ final_conf) {
  const int TILES = (PPIX + 15) / 16;    /* 1191 */
  int gid = blockIdx.x * blockDim.x + threadIdx.x;
  int wave = gid >> 5, lane = gid & 31;
  if (wave >= B_ * TILES) return;        /* wave-uniform: EXEC stays all-1s for WMMA */
  int b = wave / TILES, tile = wave % TILES;
  int base = tile * 16;
  int m = lane & 15;
  bool lo = (lane < 16);

  /* A fragment: lane m holds row M=m; lo lanes K {0..7,16..23}, hi lanes K {8..15,24..31} */
  int pixA = base + m; pixA = (pixA < PPIX) ? pixA : (PPIX - 1);
  const float* pr = proto + ((size_t)b * PPIX + pixA) * M_;
  int kb = lo ? 0 : 8;
  v16h a, bf;
  #pragma unroll
  for (int e = 0; e < 8; ++e) {
    a[e]     = (_Float16)pr[kb + e];
    a[8 + e] = (_Float16)pr[16 + kb + e];
  }
  /* B fragment: lane holds column N=m; lo lanes K {0..7,16..23}, hi lanes K {8..15,24..31} */
  const float* km = kmaskw + ((size_t)b * 16 + m) * M_;
  int ko = lo ? 0 : 8;
  #pragma unroll
  for (int e = 0; e < 8; ++e) { bf[e] = (_Float16)km[ko + e]; bf[8 + e] = (_Float16)km[16 + ko + e]; }

  v8f c = {};
  v8f d = __builtin_amdgcn_wmma_f32_16x16x32_f16(false, a, false, bf,
                                                 (short)0, c, false, false);

  /* epilogue: lane owns k = m; VGPR v owns pixel M = v (+8 for hi lanes) */
  float cx = kcx[b * 16 + m], cy = kcy[b * 16 + m];
  float sx = ksx[b * 16 + m], sy = ksy[b * 16 + m];
  float cf = kconf[b * 16 + m];
  float i2sx = fast_rcp(2.f * sx * sx), i2sy = fast_rcp(2.f * sy * sy);
  bool kvalid = (m < TOPKI);
  const float invW = 1.f / (float)PW, invH = 1.f / (float)PH;

  /* pixels in a 16-wide tile span at most 2 rows of the 138-wide image */
  int y0r = base / PW, x0r = base - y0r * PW;

  float s1[8], s2[8];
  #pragma unroll
  for (int v = 0; v < 8; ++v) {
    int mm = lo ? v : v + 8;
    int pix = base + mm;
    float mc = 0.f;
    if (kvalid && pix < PPIX) {
      int x = x0r + mm, y = y0r;
      if (x >= PW) { x -= PW; y += 1; }
      float xs = ((float)x + 0.5f) * invW, ys = ((float)y + 0.5f) * invH;
      float dx = xs - cx, dy = ys - cy;
      float ug = __expf(-(dy * dy * i2sy + dx * dx * i2sx));
      float sg = fast_rcp(1.f + __expf(-d[v]));
      mc = sg * ug * cf;
    }
    s1[v] = mc; s2[v] = mc * mc;
  }
  #pragma unroll
  for (int v = 0; v < 8; ++v) {
    #pragma unroll
    for (int msk = 8; msk >= 1; msk >>= 1) {   /* reduce over k within 16-lane group */
      s1[v] += __shfl_xor(s1[v], msk, 32);
      s2[v] += __shfl_xor(s2[v], msk, 32);
    }
  }
  if ((lane & 15) == 0) {
    #pragma unroll
    for (int v = 0; v < 8; ++v) {
      int mm = lo ? v : v + 8;
      int pix = base + mm;
      if (pix < PPIX) {
        float fc = 1.f - s2[v] / (s1[v] + EPSF);
        if (fc != fc) fc = 0.f;
        final_conf[(size_t)b * PPIX + pix] = fc;
      }
    }
  }
}

/* ---------------- 7. bilinear resize 138 -> 550 (jax half-pixel, edge clamp) ---------------- */
__global__ void k_resize(const float* __restrict__ final_conf, float* __restrict__ resized) {
  int gid = blockIdx.x * blockDim.x + threadIdx.x;
  if (gid >= B_ * OPIX) return;
  int b = gid / OPIX, p = gid % OPIX;
  int y = p / OW, x = p % OW;
  const float sc = (float)PH / (float)OH;
  float syf = (y + 0.5f) * sc - 0.5f, sxf = (x + 0.5f) * sc - 0.5f;
  int y0 = (int)floorf(syf), x0 = (int)floorf(sxf);
  float fy = syf - (float)y0, fx = sxf - (float)x0;
  int y0c = min(max(y0, 0), PH - 1), y1c = min(max(y0 + 1, 0), PH - 1);
  int x0c = min(max(x0, 0), PW - 1), x1c = min(max(x0 + 1, 0), PW - 1);
  const float* f = final_conf + (size_t)b * PPIX;
  float v00 = f[y0c * PW + x0c], v01 = f[y0c * PW + x1c];
  float v10 = f[y1c * PW + x0c], v11 = f[y1c * PW + x1c];
  resized[gid] = v00 * (1.f - fy) * (1.f - fx) + v01 * (1.f - fy) * fx
               + v10 * fy * (1.f - fx)         + v11 * fy * fx;
}

/* ---------------- 8. weighted variance, per-block partial sums (exact IEEE path) ---------------- */
__global__ void k_var(const float* __restrict__ orig, const float* __restrict__ resized,
                      float* __restrict__ partials) {
  __shared__ float sm[256];
  int tid = threadIdx.x;
  int id = blockIdx.x * 256 + tid;
  float acc = 0.f;
  if (id < OPIX) {
    __builtin_prefetch(orig + id, 0, 0);   /* global_prefetch_b8 */
    float r[4];
    #pragma unroll
    for (int bb = 0; bb < 4; ++bb) r[bb] = resized[(size_t)bb * OPIX + id];
    float total = r[0] + r[1] + r[2] + r[3];
    float inv = 1.f / (total + EPSF);
    #pragma unroll
    for (int cc = 0; cc < 3; ++cc) {
      float o[4]; float wm = 0.f;
      #pragma unroll
      for (int bb = 0; bb < 4; ++bb) {
        o[bb] = orig[((size_t)bb * 3 + cc) * OPIX + id];
        wm += o[bb] * r[bb];
      }
      #pragma unroll
      for (int bb = 0; bb < 4; ++bb) {
        float dd = o[bb] - wm;
        acc += dd * dd * r[bb];
      }
    }
    acc *= inv;
  }
  sm[tid] = acc;
  __syncthreads();
  for (int s = 128; s > 0; s >>= 1) {
    if (tid < s) sm[tid] += sm[tid + s];
    __syncthreads();
  }
  if (tid == 0) partials[blockIdx.x] = sm[0];
}

/* ---------------- 9. deterministic final reduction ---------------- */
__global__ void k_final(const float* __restrict__ partials, int n, float* __restrict__ out) {
  __shared__ float sm[1024];
  int tid = threadIdx.x;
  float a = 0.f;
  for (int i = tid; i < n; i += 1024) a += partials[i];
  sm[tid] = a;
  __syncthreads();
  for (int s = 512; s > 0; s >>= 1) {
    if (tid < s) sm[tid] += sm[tid + s];
    __syncthreads();
  }
  if (tid == 0) out[0] = sm[0] * 0.01f;
}

extern "C" void kernel_launch(void* const* d_in, const int* in_sizes, int n_in,
                              void* d_out, int out_size, void* d_ws, size_t ws_size,
                              hipStream_t stream) {
  const float* original = (const float*)d_in[0];
  const float* loc      = (const float*)d_in[1];
  const float* conf     = (const float*)d_in[2];
  const float* mask     = (const float*)d_in[3];
  const float* proto    = (const float*)d_in[4];

  float*    ws      = (float*)d_ws;        /* needs ~6.9 MB of scratch */
  float*    conf1   = ws + OFF_CONF1;
  int*      tki     = (int*)(ws + OFF_TOPKI);
  float*    tkv     = ws + OFF_TOPKV;
  float*    sloc    = ws + OFF_SLOC;
  float*    smask   = ws + OFF_SMASK;
  float*    gauss   = ws + OFF_GAUSS;
  unsigned* ioumax  = (unsigned*)(ws + OFF_IOUMAX);
  float*    kcx     = ws + OFF_KCX;
  float*    kcy     = ws + OFF_KCY;
  float*    ksx     = ws + OFF_KSX;
  float*    ksy     = ws + OFF_KSY;
  float*    kconf   = ws + OFF_KCONF;
  float*    kmaskw  = ws + OFF_KMASKW;
  float*    fconf   = ws + OFF_FINAL;
  float*    resized = ws + OFF_RESIZED;
  float*    part    = ws + OFF_PART;
  float*    out     = (float*)d_out;

  /* 1. softmax class-1: one wave per (b,p) row */
  {
    int waves = B_ * P_;
    int blocks = (waves * 32 + 255) / 256;
    k_conf1<<<blocks, 256, 0, stream>>>(conf, conf1);
  }
  /* 2. top-100 per batch */
  k_topk<<<B_, 1024, 0, stream>>>(conf1, tki, tkv);
  /* 3. 32x32 gaussians + gathers */
  k_gauss<<<B_ * TOPKC, 256, 0, stream>>>(loc, mask, tki, sloc, smask, gauss, ioumax);
  /* 4. pairwise IoU max over i<j */
  {
    int waves = B_ * (TOPKC * (TOPKC - 1) / 2);
    int blocks = (waves * 32 + 255) / 256;
    k_iou<<<blocks, 256, 0, stream>>>(gauss, ioumax);
  }
  /* 5. keep-15 + operand staging */
  k_keep<<<B_, 512, 0, stream>>>(ioumax, tkv, sloc, smask, kcx, kcy, ksx, ksy, kconf, kmaskw);
  /* 6. WMMA assemble + fused final_conf */
  {
    int tiles = (PPIX + 15) / 16;
    int waves = B_ * tiles;
    int blocks = (waves * 32 + 255) / 256;
    k_assemble<<<blocks, 256, 0, stream>>>(proto, kmaskw, kcx, kcy, ksx, ksy, kconf, fconf);
  }
  /* 7. bilinear resize */
  {
    int n = B_ * OPIX;
    k_resize<<<(n + 255) / 256, 256, 0, stream>>>(fconf, resized);
  }
  /* 8. weighted variance partials */
  {
    int blocks = (OPIX + 255) / 256;     /* 1182 */
    k_var<<<blocks, 256, 0, stream>>>(original, resized, part);
    /* 9. deterministic final sum -> scalar / 100 */
    k_final<<<1, 1024, 0, stream>>>(part, blocks, out);
  }
}
